// CausalSelfAttention_86036784873909
// MI455X (gfx1250) — matrix operand order repaired and checked by reading, not compiled, not run on hardware
//
#include <hip/hip_runtime.h>
#include <hip/hip_bf16.h>

// ---------------------------------------------------------------------------
// Causal self-attention forward, bf16 WMMA pipeline for gfx1250 (MI455X).
// B=2, S=2048, D=1024, H=16, HD=64.
// GEMMs: 64x32 wave tiles (8 WMMA / 32-K step), weights async-staged to LDS
// with global_load_async_to_lds_b128 + s_wait_asynccnt double buffering.
// ---------------------------------------------------------------------------

typedef __attribute__((ext_vector_type(16))) __bf16 v16bf;
typedef __attribute__((ext_vector_type(8)))  __bf16 v8bf;
typedef __attribute__((ext_vector_type(8)))  float  v8f;

constexpr int B_  = 2;
constexpr int S_  = 2048;
constexpr int D_  = 1024;
constexpr int H_  = 16;
constexpr int HD_ = 64;
constexpr int M_  = B_ * S_;       // 4096 rows for all projections
constexpr int KB_ = D_ / 32;       // 32 K-steps per GEMM

#if defined(__has_builtin)
#if __has_builtin(__builtin_amdgcn_s_wait_asynccnt)
#define WAIT_ASYNC(n) __builtin_amdgcn_s_wait_asynccnt(n)
#endif
#endif
#ifndef WAIT_ASYNC
#define WAIT_ASYNC(n) asm volatile("s_wait_asynccnt " #n ::: "memory")
#endif

// Join two 8-element chunks into one 16-element WMMA operand.
__device__ __forceinline__ v16bf join8(const __bf16* p0, const __bf16* p1) {
  v8bf a = *(const v8bf*)p0;
  v8bf b = *(const v8bf*)p1;
  v16bf r;
#pragma unroll
  for (int i = 0; i < 8; ++i) { r[i] = a[i]; r[i + 8] = b[i]; }
  return r;
}

__device__ __forceinline__ v8f wmma_bf16(v16bf a, v16bf b, v8f c) {
  return __builtin_amdgcn_wmma_f32_16x16x32_bf16(false, a, false, b,
                                                 (short)0, c, false, false);
}

// ---------------------------------------------------------------------------
// Converters / packers
// ---------------------------------------------------------------------------
__global__ void f32_to_bf16(const float* __restrict__ in,
                            __bf16* __restrict__ out, int n) {
  int i = blockIdx.x * blockDim.x + threadIdx.x;
  if (i < n) out[i] = (__bf16)in[i];
}

// Pack a row-major [D][D] f32 weight into WMMA B-fragment layout:
//   Wp[k/32][n/16][lane][e]  where lane = (n%16) + 16*((k%32)/16), e = k%16.
// After packing, a lane's v16bf B operand is 32 contiguous bytes, and a
// workgroup's per-K-step tile (4 n-tiles) is 4KB contiguous -> async copy.
__global__ void pack_weight(const float* __restrict__ W,
                            __bf16* __restrict__ Wp) {
  int t = blockIdx.x * blockDim.x + threadIdx.x;   // over D*D
  int k = t >> 10;            // / 1024
  int n = t & 1023;
  int kb = k >> 5, kw = k & 31, grp = kw >> 4, e = kw & 15;
  int nt = n >> 4;
  int L  = (n & 15) + 16 * grp;
  const int NT = D_ >> 4;     // 64
  Wp[((((size_t)kb * NT) + nt) * 32 + L) * 16 + e] = (__bf16)W[t];
}

// ---------------------------------------------------------------------------
// GEMM: C[M][N] = A[M][K](bf16 row-major) * Bp(packed) + bias, M=4096, N=K=1024
// 8 waves as 4(M) x 2(N); wave tile 64x32 (4x2 WMMA); WG tile 256x64.
// Weights double-buffered in LDS via async-to-LDS.
// mode 0: scatter bf16 into [b,h,s,hd]           (Q, K)
// mode 1: scatter bf16 into AV B-fragment layout  (V)
// mode 2: store f32 row-major                     (final output)
// ---------------------------------------------------------------------------
__global__ __launch_bounds__(256)
void gemm_wmma(const __bf16* __restrict__ A, const __bf16* __restrict__ Bp,
               const float* __restrict__ bias, void* __restrict__ out,
               int mode) {
  __shared__ __align__(16) __bf16 bsh[2][2048];   // 2 x 4KB B-tile staging

  const int tid  = threadIdx.x;
  const int lane = tid & 31;
  const int wave = tid >> 5;
  const int wm = wave >> 1, wn = wave & 1;
  const int m0 = blockIdx.x * 256 + wm * 64;
  const int n0 = blockIdx.y * 64 + wn * 32;
  const int nt0 = blockIdx.y * 4;                 // first n-tile of this WG
  const int g = lane >> 4, lr = lane & 15;
  const int K = D_, N = D_;
  const int NT = N >> 4;

  // Cooperative async stage: 4KB contiguous, 256 threads x 16B each.
  auto stage = [&](int kb, int buf) {
    const __bf16* src = Bp + ((size_t)kb * NT + nt0) * 512 + (size_t)tid * 8;
    // low 32 bits of a generic pointer to LDS are the LDS byte address
    unsigned ldsoff = (unsigned)(uintptr_t)(void*)&bsh[buf][tid * 8];
    asm volatile("global_load_async_to_lds_b128 %0, %1, off"
                 :: "v"(ldsoff), "v"(src) : "memory");
  };

  v8f acc[4][2] = {};
  stage(0, 0);
  for (int kb = 0; kb < KB_; ++kb) {
    const int cur = kb & 1;
    if (kb + 1 < KB_) {
      stage(kb + 1, cur ^ 1);
      WAIT_ASYNC(1);            // async completes in order: current buf ready
    } else {
      WAIT_ASYNC(0);
    }
    __syncthreads();

    v16bf af[4], bfr[2];
#pragma unroll
    for (int mi = 0; mi < 4; ++mi) {
      const __bf16* pa = A + (size_t)(m0 + mi * 16 + lr) * K + kb * 32 + g * 16;
      af[mi] = join8(pa, pa + 8);
      __builtin_prefetch(pa + 32, 0, 0);          // next K-step A row
    }
#pragma unroll
    for (int ni = 0; ni < 2; ++ni) {
      const __bf16* pb = &bsh[cur][((wn * 2 + ni) * 32 + lane) * 16];
      bfr[ni] = join8(pb, pb + 8);
    }
#pragma unroll
    for (int mi = 0; mi < 4; ++mi)
#pragma unroll
      for (int ni = 0; ni < 2; ++ni)
        acc[mi][ni] = wmma_bf16(af[mi], bfr[ni], acc[mi][ni]);
    __syncthreads();            // all reads of buf done before its reuse
  }

#pragma unroll
  for (int mi = 0; mi < 4; ++mi)
#pragma unroll
    for (int ni = 0; ni < 2; ++ni)
#pragma unroll
      for (int r = 0; r < 8; ++r) {
        int row = m0 + mi * 16 + r + 8 * g;      // global token row
        int col = n0 + ni * 16 + lr;             // global feature col
        float v = acc[mi][ni][r] + bias[col];
        if (mode == 2) {
          ((float*)out)[(size_t)row * N + col] = v;
        } else {
          int b = row / S_, s = row % S_;
          int h = col >> 6, d = col & 63;        // HD_=64
          if (mode == 0) {
            ((__bf16*)out)[(((size_t)(b * H_ + h) * S_) + s) * HD_ + d] =
                (__bf16)v;
          } else {
            // V -> AV B-fragment layout: key=s, dim=d
            int kt = s >> 5, kw = s & 31, grp = kw >> 4, e = kw & 15;
            int ntile = d >> 4;
            int L = (d & 15) + 16 * grp;
            ((__bf16*)out)[((((size_t)(b * H_ + h) * (S_ >> 5) + kt) *
                                (HD_ >> 4) + ntile) * 32 + L) * 16 + e] =
                (__bf16)v;
          }
        }
      }
}

// ---------------------------------------------------------------------------
// Flash attention: one wave per 16-query tile; 32-key blocks up to diagonal.
// Qh,Kh: [b*H+h][s][64] bf16 row-major; Vp: packed B-fragments; Ob: [B*S][D].
// ---------------------------------------------------------------------------
__global__ __launch_bounds__(128)
void attn_wmma(const __bf16* __restrict__ Qh, const __bf16* __restrict__ Kh,
               const __bf16* __restrict__ Vp, __bf16* __restrict__ Ob) {
  __shared__ __align__(16) __bf16 lds[4][16][40];  // per-wave 16x32 P tile

  const int lane = threadIdx.x & 31;
  const int wave = threadIdx.x >> 5;
  const int g = lane >> 4, lr = lane & 15;
  const int bh = blockIdx.x;                 // b*H + h
  const int b = bh >> 4, h = bh & 15;
  const int q0 = (blockIdx.y * 4 + wave) * 16;

  // Q A-fragments: two 32-dim chunks of the 64-dim head.
  const __bf16* Qbase = Qh + ((size_t)bh * S_ + q0) * HD_;
  v16bf qf[2];
#pragma unroll
  for (int c = 0; c < 2; ++c) {
    const __bf16* p = Qbase + lr * HD_ + c * 32 + g * 16;
    qf[c] = join8(p, p + 8);
  }

  float mrow[8], lrow[8];
#pragma unroll
  for (int r = 0; r < 8; ++r) { mrow[r] = -1e30f; lrow[r] = 0.f; }
  v8f oacc[4] = {};

  const int nkb = ((q0 + 15) >> 5) + 1;
  for (int kb = 0; kb < nkb; ++kb) {
    const int key0 = kb * 32;
    const bool has1 = (key0 + 16) <= (q0 + 15);

    // ---- scores: S = Q * K^T (row-major K rows are B-fragments directly)
    v8f s0 = {}, s1 = {};
    {
      const __bf16* kp = Kh + ((size_t)bh * S_ + key0 + lr) * HD_ + g * 16;
      s0 = wmma_bf16(qf[0], join8(kp, kp + 8), s0);
      s0 = wmma_bf16(qf[1], join8(kp + 32, kp + 40), s0);
    }
    if (has1) {
      const __bf16* kp = Kh + ((size_t)bh * S_ + key0 + 16 + lr) * HD_ + g * 16;
      s1 = wmma_bf16(qf[0], join8(kp, kp + 8), s1);
      s1 = wmma_bf16(qf[1], join8(kp + 32, kp + 40), s1);
    }

    // ---- online softmax over the 16x32 block
#pragma unroll
    for (int r = 0; r < 8; ++r) {
      const int qrow = q0 + r + 8 * g;
      float a0 = s0[r] * 0.125f;                       // 1/sqrt(64)
      if (key0 + lr > qrow) a0 = -1e30f;
      float a1 = -1e30f;
      if (has1) {
        a1 = s1[r] * 0.125f;
        if (key0 + 16 + lr > qrow) a1 = -1e30f;
      }
      float rm = fmaxf(a0, a1);
#pragma unroll
      for (int off = 8; off >= 1; off >>= 1)
        rm = fmaxf(rm, __shfl_xor(rm, off, 32));       // within 16-lane half
      const float mn = fmaxf(mrow[r], rm);
      const float corr = __expf(mrow[r] - mn);
      const float p0 = __expf(a0 - mn);
      const float p1 = __expf(a1 - mn);
      float rs = p0 + p1;
#pragma unroll
      for (int off = 8; off >= 1; off >>= 1)
        rs += __shfl_xor(rs, off, 32);
      lrow[r] = lrow[r] * corr + rs;
      mrow[r] = mn;
#pragma unroll
      for (int nt = 0; nt < 4; ++nt) oacc[nt][r] *= corr;
      lds[wave][r + 8 * g][lr] = (__bf16)p0;           // C-layout -> LDS
      lds[wave][r + 8 * g][16 + lr] = (__bf16)p1;
    }

    // ---- P as A-fragment from LDS (wave-private; DScnt handled by compiler)
    const __bf16* prow = &lds[wave][lr][g * 16];
    v16bf pf = join8(prow, prow + 8);

    // ---- O += P * V (V pre-swizzled into B-fragments: 2 contiguous loads)
    const __bf16* vb =
        Vp + (((size_t)bh * (S_ >> 5) + kb) * 4) * 512 + (size_t)lane * 16;
#pragma unroll
    for (int nt = 0; nt < 4; ++nt) {
      const __bf16* pv = vb + nt * 512;
      oacc[nt] = wmma_bf16(pf, join8(pv, pv + 8), oacc[nt]);
    }
  }

  // ---- normalize and merge heads back to [B*S][D]
#pragma unroll
  for (int nt = 0; nt < 4; ++nt)
#pragma unroll
    for (int r = 0; r < 8; ++r) {
      int s = q0 + r + 8 * g;
      int d = nt * 16 + lr;
      float v = oacc[nt][r] / lrow[r];
      Ob[((size_t)(b * S_) + s) * D_ + h * HD_ + d] = (__bf16)v;
    }
}

// ---------------------------------------------------------------------------
// Host launcher
// ---------------------------------------------------------------------------
extern "C" void kernel_launch(void* const* d_in, const int* in_sizes, int n_in,
                              void* d_out, int out_size, void* d_ws,
                              size_t ws_size, hipStream_t stream) {
  (void)in_sizes; (void)n_in; (void)out_size; (void)ws_size;
  const float* x  = (const float*)d_in[0];
  const float* Wq = (const float*)d_in[1];
  const float* bq = (const float*)d_in[2];
  const float* Wk = (const float*)d_in[3];
  const float* bk = (const float*)d_in[4];
  const float* Wv = (const float*)d_in[5];
  const float* bv = (const float*)d_in[6];
  const float* Wo = (const float*)d_in[7];
  const float* bo = (const float*)d_in[8];

  char* ws = (char*)d_ws;
  size_t off = 0;
  auto take = [&](size_t elems) -> __bf16* {
    __bf16* p = (__bf16*)(ws + off);
    off += (elems * 2 + 255) & ~(size_t)255;
    return p;
  };
  __bf16* xb  = take((size_t)M_ * D_);
  __bf16* Wqp = take((size_t)D_ * D_);
  __bf16* Wkp = take((size_t)D_ * D_);
  __bf16* Wvp = take((size_t)D_ * D_);
  __bf16* Wop = take((size_t)D_ * D_);
  __bf16* Qh  = take((size_t)M_ * D_);
  __bf16* Kh  = take((size_t)M_ * D_);
  __bf16* Vp  = take((size_t)M_ * D_);
  __bf16* Ob  = take((size_t)M_ * D_);

  f32_to_bf16<<<dim3((M_ * D_) / 256), 256, 0, stream>>>(x, xb, M_ * D_);
  pack_weight<<<dim3((D_ * D_) / 256), 256, 0, stream>>>(Wq, Wqp);
  pack_weight<<<dim3((D_ * D_) / 256), 256, 0, stream>>>(Wk, Wkp);
  pack_weight<<<dim3((D_ * D_) / 256), 256, 0, stream>>>(Wv, Wvp);
  pack_weight<<<dim3((D_ * D_) / 256), 256, 0, stream>>>(Wo, Wop);

  dim3 ggrid(M_ / 256, D_ / 64);
  gemm_wmma<<<ggrid, 256, 0, stream>>>(xb, Wqp, bq, (void*)Qh, 0);
  gemm_wmma<<<ggrid, 256, 0, stream>>>(xb, Wkp, bk, (void*)Kh, 0);
  gemm_wmma<<<ggrid, 256, 0, stream>>>(xb, Wvp, bv, (void*)Vp, 1);

  attn_wmma<<<dim3(B_ * H_, S_ / 64), 128, 0, stream>>>(Qh, Kh, Vp, Ob);

  gemm_wmma<<<ggrid, 256, 0, stream>>>(Ob, Wop, bo, d_out, 2);
}
